// _MultiheadAttention_5884105195931
// MI455X (gfx1250) — compile-verified, matrix operand order checked by reading, not run on hardware
//
#include <hip/hip_runtime.h>
#include <hip/hip_bf16.h>

typedef __attribute__((ext_vector_type(16))) __bf16 v16bf;
typedef __attribute__((ext_vector_type(8)))  float  v8f;
typedef int v4i __attribute__((vector_size(16)));

#define AS1 __attribute__((address_space(1)))
#define AS3 __attribute__((address_space(3)))

#define D_MODEL  1024
#define N_HEADS  16
#define HEAD_DIM 64
#define SLEN     2048
#define BATCH    2
#define MROWS    (BATCH * SLEN)   /* 4096 */

// Async global->LDS path (CDNA5 ASYNCcnt engine), guarded so compile never regresses.
#if defined(__HIP_DEVICE_COMPILE__) && defined(__gfx1250__) && \
    __has_builtin(__builtin_amdgcn_global_load_async_to_lds_b128)
#define USE_ASYNC 1
#else
#define USE_ASYNC 0
#endif

union FragBF { v16bf v; unsigned int u[8]; };

__device__ __forceinline__ unsigned int pack2bf(float a, float b) {
    // bf16 truncation pack: low 16 = bf16(a), high 16 = bf16(b)
    unsigned int ua = __builtin_bit_cast(unsigned int, a);
    unsigned int ub = __builtin_bit_cast(unsigned int, b);
    return (ua >> 16) | (ub & 0xFFFF0000u);
}

template <int N>
__device__ __forceinline__ void wait_async() {
#if USE_ASYNC
#if __has_builtin(__builtin_amdgcn_s_wait_asynccnt)
    __builtin_amdgcn_s_wait_asynccnt(N);
#else
    asm volatile("s_wait_asynccnt %0" :: "i"(N));
#endif
#endif
}

// A tile: 64 rows x 32 k, fp32, row stride 36 (16B-aligned segments, bank spread)
// B tile: 32 k x 64 n, fp32, raw row-major as in global memory
#define A_STRIDE 36

__device__ __forceinline__ void issue_tile(const float* __restrict__ X,
                                           const float* __restrict__ W,
                                           int bm, int bn, int k0,
                                           float* __restrict__ As,
                                           float* __restrict__ Bs, int tid)
{
#pragma unroll
    for (int s = 0; s < 2; ++s) {               // A: 512 x 16B segments
        const int idx = (tid + s * 256) * 4;
        const int m = idx >> 5, kk = idx & 31;
        const float* g = X + (size_t)(bm + m) * D_MODEL + k0 + kk;
        float* l = As + m * A_STRIDE + kk;
#if USE_ASYNC
        __builtin_amdgcn_global_load_async_to_lds_b128(
            (AS1 v4i*)(void*)g, (AS3 v4i*)(void*)l, 0, 0);
#else
        *(float4*)l = *(const float4*)g;
#endif
    }
#pragma unroll
    for (int s = 0; s < 2; ++s) {               // B: 512 x 16B segments
        const int idx = (tid + s * 256) * 4;
        const int k = idx >> 6, n = idx & 63;
        const float* g = W + (size_t)(k0 + k) * D_MODEL + bn + n;
        float* l = Bs + k * 64 + n;
#if USE_ASYNC
        __builtin_amdgcn_global_load_async_to_lds_b128(
            (AS1 v4i*)(void*)g, (AS3 v4i*)(void*)l, 0, 0);
#else
        *(float4*)l = *(const float4*)g;
#endif
    }
}

// ---------------------------------------------------------------------------
// GEMM: C[4096,1024] = X[4096,1024] @ W[1024,1024] + bias, bf16 WMMA, fp32 acc.
// Double-buffered async global->LDS staging (raw fp32), fp32->bf16 pack during
// fragment build. Block: 256 threads (8 waves), 64x64 tile, K-step 32.
// MODE 0: out bf16 [b,h,s,d] (Q)   MODE 1: bf16 [b,h,s,d] (K)
// MODE 2: out bf16 [b,h,d,s] (V^T) MODE 3: fp32 [row,col] (O -> d_out)
// ---------------------------------------------------------------------------
template <int MODE>
__global__ __launch_bounds__(256)
void gemm_proj_kernel(const float* __restrict__ X, const float* __restrict__ W,
                      const float* __restrict__ bias, void* __restrict__ outp)
{
    __shared__ float As_f[2][64 * A_STRIDE];
    __shared__ float Bs_f[2][32 * 64];

    const int tid   = threadIdx.x;
    const int lane  = tid & 31;
    const int wid   = tid >> 5;
    const int waveM = wid & 1;    // 2 waves along M
    const int waveN = wid >> 1;   // 4 waves along N
    const int bm = blockIdx.y * 64;
    const int bn = blockIdx.x * 64;

    v8f acc[2];
    acc[0] = (v8f){0.f,0.f,0.f,0.f,0.f,0.f,0.f,0.f};
    acc[1] = (v8f){0.f,0.f,0.f,0.f,0.f,0.f,0.f,0.f};

    const int kb = (lane >> 4) * 8;   // fragment K base per half-wave

    issue_tile(X, W, bm, bn, 0, As_f[0], Bs_f[0], tid);

    for (int step = 0; step < D_MODEL / 32; ++step) {
        const int cur = step & 1;
        if (step + 1 < D_MODEL / 32) {
            issue_tile(X, W, bm, bn, (step + 1) * 32, As_f[1 - cur], Bs_f[1 - cur], tid);
            wait_async<4>();   // previous tile's 4 async ops retired (in-order)
        } else {
            wait_async<0>();
        }
        __syncthreads();

        const float* As = As_f[cur];
        const float* Bs = Bs_f[cur];

        FragBF bfr;
        {
            const int n = waveN * 16 + (lane & 15);
            #pragma unroll
            for (int v = 0; v < 8; ++v) {
                const int k = kb + (v < 4 ? 2 * v : 16 + 2 * (v - 4));
                bfr.u[v] = pack2bf(Bs[k * 64 + n], Bs[(k + 1) * 64 + n]);
            }
        }
        #pragma unroll
        for (int mt = 0; mt < 2; ++mt) {
            FragBF afr;
            const int m = waveM * 32 + mt * 16 + (lane & 15);
            #pragma unroll
            for (int v = 0; v < 8; ++v) {
                const int k = kb + (v < 4 ? 2 * v : 16 + 2 * (v - 4));
                const float2 f2 = *(const float2*)&As[m * A_STRIDE + k];
                afr.u[v] = pack2bf(f2.x, f2.y);
            }
            acc[mt] = __builtin_amdgcn_wmma_f32_16x16x32_bf16(
                false, afr.v, false, bfr.v, (short)0, acc[mt], false, false);
        }
        __syncthreads();   // all waves done reading `cur` before it is overwritten
    }

    // epilogue: C layout = VGPR r, lanes 0-15 -> M=r, lanes 16-31 -> M=8+r
    const int lrow = (lane >> 4) * 8;
    const int lcol = lane & 15;
    #pragma unroll
    for (int mt = 0; mt < 2; ++mt) {
        #pragma unroll
        for (int r = 0; r < 8; ++r) {
            const int grow = bm + waveM * 32 + mt * 16 + lrow + r;
            const int gcol = bn + waveN * 16 + lcol;
            const float val = acc[mt][r] + bias[gcol];
            if (MODE == 3) {
                ((float*)outp)[(size_t)grow * D_MODEL + gcol] = val;
            } else {
                const int h = gcol >> 6, d = gcol & 63;
                const int bb = grow >> 11, s = grow & 2047;
                const __hip_bfloat16 bv16 = __float2bfloat16(val);
                if (MODE == 2)  // transposed [b,h,d,s]
                    ((__hip_bfloat16*)outp)[((size_t)(bb*N_HEADS + h)*HEAD_DIM + d)*SLEN + s] = bv16;
                else            // [b,h,s,d]
                    ((__hip_bfloat16*)outp)[((size_t)(bb*N_HEADS + h)*SLEN + s)*HEAD_DIM + d] = bv16;
            }
        }
    }
}

// ---------------------------------------------------------------------------
// Attention: one block per (b, h, 32-row q tile). 8 waves; wave w owns key
// columns [w*256, w*256+256). 32x2048 fp32 score tile lives in LDS (256KB,
// fits CDNA5's 320KB/WGP). Softmax in fp32; probs stored to global (fp32
// attn_weights output) and consumed from LDS (packed to bf16) for P@V.
// ---------------------------------------------------------------------------
__global__ __launch_bounds__(256)
void attention_kernel(const __hip_bfloat16* __restrict__ qb,
                      const __hip_bfloat16* __restrict__ kb,
                      const __hip_bfloat16* __restrict__ vt,
                      float* __restrict__ attn_out,
                      float* __restrict__ ctx,
                      const float* __restrict__ scale_p)
{
    extern __shared__ char smem[];
    float* S    = (float*)smem;                                // 32*2048 fp32
    float* red  = (float*)(smem + 32 * 2048 * 4);              // 32*8
    float* Obuf = (float*)(smem + 32 * 2048 * 4 + 32 * 8 * 4); // 32*64

    const int tid  = threadIdx.x;
    const int lane = tid & 31;
    const int wid  = tid >> 5;
    const int qt = blockIdx.x;   // 0..63 (32-row q tiles)
    const int h  = blockIdx.y;
    const int b  = blockIdx.z;
    const float scale = scale_p[0];

    const size_t headrow = (size_t)(b * N_HEADS + h) * SLEN;

    for (int i = tid; i < 32 * 64; i += 256) Obuf[i] = 0.f;

    const unsigned int* qu = (const unsigned int*)qb;  // 32 dwords per 64-d row
    const unsigned int* ku = (const unsigned int*)kb;
    const unsigned int* vu = (const unsigned int*)vt;  // 1024 dwords per 2048-s row

    const int kq = (lane >> 4) * 4;

    // preload q fragments: 2 m-tiles x 2 d-halves
    FragBF qf[2][2];
    #pragma unroll
    for (int mt = 0; mt < 2; ++mt) {
        const int row = qt * 32 + mt * 16 + (lane & 15);
        const size_t base = (headrow + row) * 32;
        #pragma unroll
        for (int dk = 0; dk < 2; ++dk)
            #pragma unroll
            for (int v = 0; v < 8; ++v)
                qf[mt][dk].u[v] = qu[base + dk * 16 + kq + (v < 4 ? v : v + 4)];
    }

    // ---- phase 1: S = (q @ k^T) * scale ----
    const int c0w = wid * 256;
    for (int t = 0; t < 16; ++t) {
        const int c0 = c0w + t * 16;
        FragBF kf[2];
        {
            const int kpos = c0 + (lane & 15);
            const size_t base = (headrow + kpos) * 32;
            #pragma unroll
            for (int dk = 0; dk < 2; ++dk)
                #pragma unroll
                for (int v = 0; v < 8; ++v)
                    kf[dk].u[v] = ku[base + dk * 16 + kq + (v < 4 ? v : v + 4)];
        }
        #pragma unroll
        for (int mt = 0; mt < 2; ++mt) {
            v8f sacc = (v8f){0.f,0.f,0.f,0.f,0.f,0.f,0.f,0.f};
            sacc = __builtin_amdgcn_wmma_f32_16x16x32_bf16(
                false, qf[mt][0].v, false, kf[0].v, (short)0, sacc, false, false);
            sacc = __builtin_amdgcn_wmma_f32_16x16x32_bf16(
                false, qf[mt][1].v, false, kf[1].v, (short)0, sacc, false, false);
            const int lrow = (lane >> 4) * 8, lcol = lane & 15;
            #pragma unroll
            for (int r = 0; r < 8; ++r)
                S[(mt * 16 + lrow + r) * 2048 + c0 + lcol] = sacc[r] * scale;
        }
    }
    __syncthreads();

    // ---- phase 2: softmax over 2048, 8 threads per row ----
    {
        const int row = tid >> 3;
        const int sub = tid & 7;
        float* srow = S + row * 2048 + sub * 256;
        float pmax = -3.0e38f;
        for (int i = 0; i < 256; ++i) pmax = fmaxf(pmax, srow[i]);
        red[row * 8 + sub] = pmax;
        __syncthreads();
        float rmax = red[row * 8];
        #pragma unroll
        for (int i = 1; i < 8; ++i) rmax = fmaxf(rmax, red[row * 8 + i]);
        __syncthreads();
        float psum = 0.f;
        for (int i = 0; i < 256; ++i) {
            float e = __expf(srow[i] - rmax);
            srow[i] = e;
            psum += e;
        }
        red[row * 8 + sub] = psum;
        __syncthreads();
        float rsum = 0.f;
        #pragma unroll
        for (int i = 0; i < 8; ++i) rsum += red[row * 8 + i];
        const float inv = 1.f / rsum;
        float* gout = attn_out + (headrow + qt * 32 + row) * (size_t)SLEN + sub * 256;
        for (int i = 0; i < 256; ++i) {
            float p = srow[i] * inv;
            srow[i] = p;        // keep fp32 probs in LDS for P@V
            gout[i] = p;        // attn_weights output (fp32)
        }
    }
    __syncthreads();

    // ---- phase 3: O = P @ V, wave w reduces its 256-wide K slice ----
    v8f oacc[2][4];
    #pragma unroll
    for (int mt = 0; mt < 2; ++mt)
        #pragma unroll
        for (int nt = 0; nt < 4; ++nt)
            oacc[mt][nt] = (v8f){0.f,0.f,0.f,0.f,0.f,0.f,0.f,0.f};

    for (int ks = 0; ks < 8; ++ks) {
        const int kl0 = c0w + ks * 32;
        FragBF pf[2];
        {
            const int lrow = lane & 15;
            const int kb_  = (lane >> 4) * 8;
            #pragma unroll
            for (int mt = 0; mt < 2; ++mt) {
                const int row = mt * 16 + lrow;
                #pragma unroll
                for (int v = 0; v < 8; ++v) {
                    const int kk = kl0 + kb_ + (v < 4 ? 2 * v : 16 + 2 * (v - 4));
                    pf[mt].u[v] = pack2bf(S[row * 2048 + kk], S[row * 2048 + kk + 1]);
                }
            }
        }
        #pragma unroll
        for (int nt = 0; nt < 4; ++nt) {
            FragBF vf;
            const int d = nt * 16 + (lane & 15);
            const size_t vbase = ((size_t)(b * N_HEADS + h) * HEAD_DIM + d) * (SLEN / 2);
            #pragma unroll
            for (int v = 0; v < 8; ++v)
                vf.u[v] = vu[vbase + (kl0 >> 1) + kq + (v < 4 ? v : v + 4)];
            #pragma unroll
            for (int mt = 0; mt < 2; ++mt)
                oacc[mt][nt] = __builtin_amdgcn_wmma_f32_16x16x32_bf16(
                    false, pf[mt].v, false, vf.v, (short)0, oacc[mt][nt], false, false);
        }
    }

    // cross-wave reduction via LDS float atomics (ds_add_f32)
    {
        const int lrow = (lane >> 4) * 8, lcol = lane & 15;
        #pragma unroll
        for (int mt = 0; mt < 2; ++mt)
            #pragma unroll
            for (int nt = 0; nt < 4; ++nt)
                #pragma unroll
                for (int r = 0; r < 8; ++r)
                    atomicAdd(&Obuf[(mt * 16 + lrow + r) * 64 + nt * 16 + lcol],
                              oacc[mt][nt][r]);
    }
    __syncthreads();

    // write context [row, h*64+d] fp32 for the output projection
    for (int i = tid; i < 32 * 64; i += 256) {
        const int row = i >> 6, col = i & 63;
        ctx[((size_t)(b * SLEN + qt * 32 + row)) * D_MODEL + h * HEAD_DIM + col] = Obuf[i];
    }
}

// ---------------------------------------------------------------------------
extern "C" void kernel_launch(void* const* d_in, const int* in_sizes, int n_in,
                              void* d_out, int out_size, void* d_ws, size_t ws_size,
                              hipStream_t stream) {
    const float* Q     = (const float*)d_in[0];
    const float* Wq    = (const float*)d_in[1];
    const float* bq    = (const float*)d_in[2];
    const float* Wk    = (const float*)d_in[3];
    const float* bk    = (const float*)d_in[4];
    const float* Wv    = (const float*)d_in[5];
    const float* bv    = (const float*)d_in[6];
    const float* Wo    = (const float*)d_in[7];
    const float* bo    = (const float*)d_in[8];
    const float* scale = (const float*)d_in[9];

    float* outp = (float*)d_out;
    float* attn = outp + (size_t)MROWS * D_MODEL;   // attn_weights region

    char* ws = (char*)d_ws;
    __hip_bfloat16* qb  = (__hip_bfloat16*)(ws);                          // 8 MB
    __hip_bfloat16* kbp = (__hip_bfloat16*)(ws + (size_t)8  * 1024*1024); // 8 MB
    __hip_bfloat16* vt  = (__hip_bfloat16*)(ws + (size_t)16 * 1024*1024); // 8 MB
    float*          ctx = (float*)        (ws + (size_t)24 * 1024*1024);  // 16 MB

    dim3 gg(D_MODEL / 64, MROWS / 64);  // (16, 64)
    gemm_proj_kernel<0><<<gg, 256, 0, stream>>>(Q, Wq, bq, qb);
    gemm_proj_kernel<1><<<gg, 256, 0, stream>>>(Q, Wk, bk, kbp);
    gemm_proj_kernel<2><<<gg, 256, 0, stream>>>(Q, Wv, bv, vt);

    const size_t smem = (size_t)32 * 2048 * 4 + 32 * 8 * 4 + 32 * 64 * 4; // ~265 KB
    attention_kernel<<<dim3(SLEN / 32, N_HEADS, BATCH), 256, smem, stream>>>(
        qb, kbp, vt, attn, ctx, scale);

    gemm_proj_kernel<3><<<gg, 256, 0, stream>>>(ctx, Wo, bo, outp);
}
